// GNNEncoder_4389456576912
// MI455X (gfx1250) — matrix operand order repaired and checked by reading, not compiled, not run on hardware
//
#include <hip/hip_runtime.h>
#include <hip/hip_bf16.h>

#define NNODES 20000
#define NEDGES 160000
#define NGRAPH 128
#define HIDDIM 256
#define NHEAD  4
#define OUTDIM 1024
#define NLAYER 3

typedef __attribute__((ext_vector_type(16))) __bf16 v16bf;
typedef __attribute__((ext_vector_type(8)))  __bf16 v8bf;
typedef __attribute__((ext_vector_type(8)))  float  v8f;

// ---------- helpers ----------
__device__ __forceinline__ unsigned short f2bf(float f) {
    unsigned int u = __float_as_uint(f);
    unsigned int r = (u + 0x7FFFu + ((u >> 16) & 1u)) >> 16;
    return (unsigned short)r;
}

__device__ __forceinline__ void atomicMaxF(float* addr, float val) {
    if (val >= 0.f) atomicMax((int*)addr, __float_as_int(val));
    else            atomicMin((unsigned int*)addr, __float_as_uint(val));
}

__device__ __forceinline__ v16bf load_afrag(const unsigned short* p, int kb) {
    // A-fragment (16x32 bf16, ISA layout): two contiguous 8-element K-chunks
    v8bf a0 = *reinterpret_cast<const v8bf*>(p + kb);
    v8bf a1 = *reinterpret_cast<const v8bf*>(p + kb + 16);
    v16bf a;
#pragma unroll
    for (int i = 0; i < 8; ++i) { a[i] = a0[i]; a[i + 8] = a1[i]; }
    return a;
}

// ---------- feature construction: x = [type_emb | sinusoidal_pe | context[batch]] ----------
__global__ void k_build_x(const int* __restrict__ xtypes, const float* __restrict__ pos,
                          const int* __restrict__ batch, const float* __restrict__ ctx,
                          const float* __restrict__ embW, const float* __restrict__ embB,
                          float* __restrict__ x) {
    int n = blockIdx.x;
    int c = threadIdx.x;
    float v;
    if (c < 80) {
        v = embW[xtypes[n] * 80 + c] + embB[c];
    } else if (c < 128) {
        int p = c - 80;
        int coord = p >> 4;          // /16
        int j = p & 15;
        int jj = (j < 8) ? j : j - 8;
        float d = __expf(-9.2103403719761836f * (float)jj / 24.f); // 10000^(-jj/24)
        float s = pos[n * 3 + coord] * d;
        v = (j < 8) ? __sinf(s) : __cosf(s);
    } else {
        v = ctx[batch[n] * 128 + (c - 128)];
    }
    x[n * HIDDIM + c] = v;
}

// ---------- f32 -> bf16 ----------
__global__ void k_cvt_bf16(const float* __restrict__ src, unsigned short* __restrict__ dst, int n) {
    int i = blockIdx.x * blockDim.x + threadIdx.x;
    if (i < n) dst[i] = f2bf(src[i]);
}

// ---------- weight prep: transpose [256,1024] -> bf16 [1024,256] for 4 matrices ----------
__global__ void k_prep_w(const float* __restrict__ linW, const float* __restrict__ outW,
                         unsigned short* __restrict__ Wb) {
    int id = blockIdx.x * blockDim.x + threadIdx.x; // 4*1024*256
    if (id >= 4 * 1024 * 256) return;
    int m = id >> 18;            // matrix 0..3
    int r = id & 0x3FFFF;
    int n = r >> 8;              // 0..1023  (output column)
    int k = r & 255;             // 0..255   (input row)
    const float* src = (m < 3) ? (linW + (size_t)m * 256 * 1024) : outW;
    Wb[id] = f2bf(src[(size_t)k * 1024 + n]);
}

// ---------- WMMA GEMM: Y[20000,1024] = Xb[20000,256] @ W (W stored transposed bf16 [1024,256])
// One wave computes a 32x64 output tile: 2 A frags x 4 B frags -> 8 WMMAs per k-step.
__global__ void k_gemm_bf16(const unsigned short* __restrict__ Xb,
                            const unsigned short* __restrict__ Wb,
                            float* __restrict__ Y) {
    int wave = (blockIdx.x << 3) + (threadIdx.x >> 5);   // 8 waves/block, 10000 waves
    int lane = threadIdx.x & 31;
    int tm = wave >> 4;          // 0..624  (32-row tiles)
    int tn = wave & 15;          // 0..15   (64-col tiles)
    int half = lane >> 4;        // 0 | 1
    int l15 = lane & 15;

    const unsigned short* arow0 = Xb + (size_t)(tm * 32 + l15) * 256 + half * 8;
    const unsigned short* arow1 = arow0 + (size_t)16 * 256;
    const unsigned short* brow0 = Wb + (size_t)(tn * 64 + l15) * 256 + half * 8;

    v8f c00 = {}, c01 = {}, c02 = {}, c03 = {};
    v8f c10 = {}, c11 = {}, c12 = {}, c13 = {};

#pragma unroll
    for (int kb = 0; kb < 256; kb += 32) {
        __builtin_prefetch(arow0 + kb + 32, 0, 1);
        __builtin_prefetch(brow0 + kb + 32, 0, 1);
        v16bf a0 = load_afrag(arow0, kb);
        v16bf a1 = load_afrag(arow1, kb);
        v16bf b0 = load_afrag(brow0, kb);                    // col tile 0
        v16bf b1 = load_afrag(brow0 + (size_t)16 * 256, kb); // col tile 1
        v16bf b2 = load_afrag(brow0 + (size_t)32 * 256, kb); // col tile 2
        v16bf b3 = load_afrag(brow0 + (size_t)48 * 256, kb); // col tile 3
        c00 = __builtin_amdgcn_wmma_f32_16x16x32_bf16(false, a0, false, b0, (short)0, c00, false, false);
        c01 = __builtin_amdgcn_wmma_f32_16x16x32_bf16(false, a0, false, b1, (short)0, c01, false, false);
        c02 = __builtin_amdgcn_wmma_f32_16x16x32_bf16(false, a0, false, b2, (short)0, c02, false, false);
        c03 = __builtin_amdgcn_wmma_f32_16x16x32_bf16(false, a0, false, b3, (short)0, c03, false, false);
        c10 = __builtin_amdgcn_wmma_f32_16x16x32_bf16(false, a1, false, b0, (short)0, c10, false, false);
        c11 = __builtin_amdgcn_wmma_f32_16x16x32_bf16(false, a1, false, b1, (short)0, c11, false, false);
        c12 = __builtin_amdgcn_wmma_f32_16x16x32_bf16(false, a1, false, b2, (short)0, c12, false, false);
        c13 = __builtin_amdgcn_wmma_f32_16x16x32_bf16(false, a1, false, b3, (short)0, c13, false, false);
    }

    // D layout: vgpr r -> row (half*8 + r), col = l15
    float* ybase = Y + (size_t)(tm * 32 + half * 8) * 1024 + tn * 64 + l15;
#pragma unroll
    for (int r = 0; r < 8; ++r) {
        float* yr = ybase + (size_t)r * 1024;
        yr[0]  = c00[r];
        yr[16] = c01[r];
        yr[32] = c02[r];
        yr[48] = c03[r];
        float* yr2 = yr + (size_t)16 * 1024;
        yr2[0]  = c10[r];
        yr2[16] = c11[r];
        yr2[32] = c12[r];
        yr2[48] = c13[r];
    }
}

// ---------- per-node attention logits: a_src[n,h], a_dst[n,h] ----------
__global__ void k_node_att(const float* __restrict__ xl, const float* __restrict__ as,
                           const float* __restrict__ ad, float* __restrict__ asn,
                           float* __restrict__ adn) {
    int t = blockIdx.x * blockDim.x + threadIdx.x;
    if (t >= NNODES * NHEAD) return;
    int n = t >> 2, h = t & 3;
    const float* row = xl + (size_t)n * 1024 + h * 256;
    const float* sv = as + h * 256;
    const float* dv = ad + h * 256;
    float s1 = 0.f, s2 = 0.f;
    for (int c = 0; c < 256; ++c) { float v = row[c]; s1 += v * sv[c]; s2 += v * dv[c]; }
    asn[t] = s1; adn[t] = s2;
}

// ---------- collapse edge path: v[h,f] = sum_c eW[f, h*256+c] * a_e[h*256+c] ----------
__global__ void k_edge_vec(const float* __restrict__ eW, const float* __restrict__ ae,
                           float* __restrict__ vhf) {
    int t = threadIdx.x;
    if (t >= NHEAD * 5) return;
    int h = t / 5, f = t % 5;
    float s = 0.f;
    for (int c = 0; c < 256; ++c) s += eW[f * 1024 + h * 256 + c] * ae[h * 256 + c];
    vhf[h * 5 + f] = s;
}

__global__ void k_fill(float* p, float v, int n) {
    int i = blockIdx.x * blockDim.x + threadIdx.x;
    if (i < n) p[i] = v;
}

// ---------- edge alpha = leaky_relu(a_src[src]+a_dst[dst]+a_edge); segment max ----------
__global__ void k_edge_alpha(const int* __restrict__ src, const int* __restrict__ dst,
                             const float* __restrict__ eattr, const float* __restrict__ vhf,
                             const float* __restrict__ asn, const float* __restrict__ adn,
                             float* __restrict__ alpha, float* __restrict__ amax) {
    int t = blockIdx.x * blockDim.x + threadIdx.x;
    if (t >= NEDGES * NHEAD) return;
    int e = t >> 2, h = t & 3;
    int s = src[e], d = dst[e];
    float a = asn[s * 4 + h] + adn[d * 4 + h];
#pragma unroll
    for (int f = 0; f < 5; ++f) a += eattr[e * 5 + f] * vhf[h * 5 + f];
    a = (a > 0.f) ? a : 0.2f * a;
    alpha[t] = a;
    atomicMaxF(&amax[d * 4 + h], a);
}

// ---------- ex = exp(alpha - amax[dst]); segment sum ----------
__global__ void k_edge_exp(const int* __restrict__ dst, const float* __restrict__ alpha,
                           const float* __restrict__ amax, float* __restrict__ exw,
                           float* __restrict__ denom) {
    int t = blockIdx.x * blockDim.x + threadIdx.x;
    if (t >= NEDGES * NHEAD) return;
    int e = t >> 2, h = t & 3;
    int d = dst[e];
    float am = amax[d * 4 + h];
    if (!isfinite(am)) am = 0.f;
    float ex = __expf(alpha[t] - am);
    exw[t] = ex;
    atomicAdd(&denom[d * 4 + h], ex);
}

// ---------- message scatter with head-mean folded: acc[dst,c] += sum_h xl[src,h,c]*w[e,h] ----------
__global__ void k_scatter(const int* __restrict__ src, const int* __restrict__ dst,
                          const float* __restrict__ xl, const float* __restrict__ exw,
                          const float* __restrict__ denom, float* __restrict__ acc) {
    int e = blockIdx.x;
    int c = threadIdx.x;
    int s = src[e], d = dst[e];
    float m = 0.f;
#pragma unroll
    for (int h = 0; h < 4; ++h) {
        float w = exw[e * 4 + h] / (denom[d * 4 + h] + 1e-16f);
        m += xl[(size_t)s * 1024 + h * 256 + c] * w;
    }
    atomicAdd(&acc[(size_t)d * 256 + c], m);
}

// ---------- batchnorm stats (per channel over all nodes) ----------
__global__ void k_bn_stats(const float* __restrict__ acc, const float* __restrict__ cb,
                           float* __restrict__ mu, float* __restrict__ var) {
    int c = blockIdx.x;  // 256
    float s = 0.f, ss = 0.f;
    for (int n = threadIdx.x; n < NNODES; n += blockDim.x) {
        float y = acc[(size_t)n * 256 + c] * 0.25f + cb[c];
        s += y; ss += y * y;
    }
    __shared__ float sh[256], sh2[256];
    sh[threadIdx.x] = s; sh2[threadIdx.x] = ss;
    __syncthreads();
    for (int o = 128; o > 0; o >>= 1) {
        if (threadIdx.x < o) { sh[threadIdx.x] += sh[threadIdx.x + o]; sh2[threadIdx.x] += sh2[threadIdx.x + o]; }
        __syncthreads();
    }
    if (threadIdx.x == 0) {
        float m = sh[0] / (float)NNODES;
        mu[c] = m;
        var[c] = sh2[0] / (float)NNODES - m * m;
    }
}

// ---------- normalize + gelu(exact) + residual ----------
__global__ void k_bn_gelu_res(const float* __restrict__ acc, const float* __restrict__ cb,
                              const float* __restrict__ mu, const float* __restrict__ var,
                              const float* __restrict__ gamma, const float* __restrict__ beta,
                              float* __restrict__ x) {
    int n = blockIdx.x;
    int c = threadIdx.x;
    float y = acc[(size_t)n * 256 + c] * 0.25f + cb[c];
    float z = (y - mu[c]) * rsqrtf(var[c] + 1e-5f) * gamma[c] + beta[c];
    float g = 0.5f * z * (1.f + erff(z * 0.70710678118654752f));
    x[(size_t)n * 256 + c] += g;
}

// ---------- global add pool (with per-node out bias) ----------
__global__ void k_pool(const float* __restrict__ Y, const float* __restrict__ ob,
                       const int* __restrict__ batch, float* __restrict__ out) {
    int id = blockIdx.x * blockDim.x + threadIdx.x;
    if (id >= NNODES * OUTDIM) return;
    int n = id >> 10, j = id & 1023;
    atomicAdd(&out[(size_t)batch[n] * OUTDIM + j], Y[id] + ob[j]);
}

// ---------------------------------------------------------------------------
extern "C" void kernel_launch(void* const* d_in, const int* in_sizes, int n_in,
                              void* d_out, int out_size, void* d_ws, size_t ws_size,
                              hipStream_t stream) {
    const int*   x_types = (const int*)  d_in[0];
    const float* pos     = (const float*)d_in[1];
    const int*   eidx    = (const int*)  d_in[2];
    const float* eattr   = (const float*)d_in[3];
    const int*   batch   = (const int*)  d_in[4];
    const float* ctx     = (const float*)d_in[5];
    const float* embW    = (const float*)d_in[6];
    const float* embB    = (const float*)d_in[7];
    const float* linW    = (const float*)d_in[8];
    const float* attS    = (const float*)d_in[9];
    const float* attD    = (const float*)d_in[10];
    const float* edgeW   = (const float*)d_in[11];
    const float* attE    = (const float*)d_in[12];
    const float* convB   = (const float*)d_in[13];
    const float* bnG     = (const float*)d_in[14];
    const float* bnB     = (const float*)d_in[15];
    const float* outW    = (const float*)d_in[16];
    const float* outB    = (const float*)d_in[17];
    float* out = (float*)d_out;

    const int* src = eidx;
    const int* dst = eidx + NEDGES;

    // workspace carve-up (256B aligned)
    char* w = (char*)d_ws;
    size_t off = 0;
    auto alloc = [&](size_t bytes) -> void* {
        off = (off + 255) & ~(size_t)255;
        void* p = w + off;
        off += bytes;
        return p;
    };
    float*          x     = (float*)alloc((size_t)NNODES * 256 * 4);
    unsigned short* xb    = (unsigned short*)alloc((size_t)NNODES * 256 * 2);
    unsigned short* Wb    = (unsigned short*)alloc((size_t)4 * 1024 * 256 * 2);
    float*          xl    = (float*)alloc((size_t)NNODES * 1024 * 4);   // also final Y
    float*          acc   = (float*)alloc((size_t)NNODES * 256 * 4);
    float*          asn   = (float*)alloc((size_t)NNODES * 4 * 4);
    float*          adn   = (float*)alloc((size_t)NNODES * 4 * 4);
    float*          vhf   = (float*)alloc(256);
    float*          alpha = (float*)alloc((size_t)NEDGES * 4 * 4);
    float*          exw   = (float*)alloc((size_t)NEDGES * 4 * 4);
    float*          amax  = (float*)alloc((size_t)NNODES * 4 * 4);
    float*          denom = (float*)alloc((size_t)NNODES * 4 * 4);
    float*          mu    = (float*)alloc(1024);
    float*          var   = (float*)alloc(1024);

    hipMemsetAsync(d_out, 0, (size_t)NGRAPH * OUTDIM * 4, stream);

    k_prep_w<<<(4 * 1024 * 256 + 255) / 256, 256, 0, stream>>>(linW, outW, Wb);
    k_build_x<<<NNODES, 256, 0, stream>>>(x_types, pos, batch, ctx, embW, embB, x);

    for (int l = 0; l < NLAYER; ++l) {
        k_cvt_bf16<<<NNODES, 256, 0, stream>>>(x, xb, NNODES * 256);
        k_gemm_bf16<<<1250, 256, 0, stream>>>(xb, Wb + (size_t)l * 1024 * 256, xl);
        k_node_att<<<(NNODES * 4 + 255) / 256, 256, 0, stream>>>(
            xl, attS + l * 1024, attD + l * 1024, asn, adn);
        k_edge_vec<<<1, 32, 0, stream>>>(edgeW + l * 5 * 1024, attE + l * 1024, vhf);
        k_fill<<<(NNODES * 4 + 255) / 256, 256, 0, stream>>>(amax, -__builtin_huge_valf(), NNODES * 4);
        hipMemsetAsync(denom, 0, (size_t)NNODES * 4 * 4, stream);
        hipMemsetAsync(acc, 0, (size_t)NNODES * 256 * 4, stream);
        k_edge_alpha<<<(NEDGES * 4 + 255) / 256, 256, 0, stream>>>(
            src, dst, eattr, vhf, asn, adn, alpha, amax);
        k_edge_exp<<<(NEDGES * 4 + 255) / 256, 256, 0, stream>>>(dst, alpha, amax, exw, denom);
        k_scatter<<<NEDGES, 256, 0, stream>>>(src, dst, xl, exw, denom, acc);
        k_bn_stats<<<256, 256, 0, stream>>>(acc, convB + l * 256, mu, var);
        k_bn_gelu_res<<<NNODES, 256, 0, stream>>>(acc, convB + l * 256, mu, var,
                                                  bnG + l * 256, bnB + l * 256, x);
    }

    // output projection + pool
    k_cvt_bf16<<<NNODES, 256, 0, stream>>>(x, xb, NNODES * 256);
    k_gemm_bf16<<<1250, 256, 0, stream>>>(xb, Wb + (size_t)3 * 1024 * 256, xl);
    k_pool<<<(NNODES * OUTDIM + 255) / 256, 256, 0, stream>>>(xl, outB, batch, out);
}
